// Seq2Seq_32890859553310
// MI455X (gfx1250) — matrix operand order repaired and checked by reading, not compiled, not run on hardware
//
#include <hip/hip_runtime.h>
#include <hip/hip_bf16.h>

typedef __bf16 bf16_t;
typedef bf16_t v16bf __attribute__((ext_vector_type(16)));
typedef float  v8f   __attribute__((ext_vector_type(8)));

static constexpr int Bn  = 64;      // batch
static constexpr int Sn  = 64;      // src length
static constexpr int Hn  = 512;     // hidden
static constexpr int H3n = 1536;    // 3*hidden
static constexpr int Vn  = 32000;   // vocab
static constexpr int Tn  = 80;      // max_len

// ---------------- fragment loaders (contiguous-k layouts, A and B match) ----------------
// A (16x32 bf16): lane half=lane>>4, r=lane&15; elems 0..15 = K k0+half*16..+15
// (contiguous, matching B).
__device__ __forceinline__ v16bf load_frag_a(const bf16_t* p) {
  v16bf f;
  uint4* u = reinterpret_cast<uint4*>(&f);
  u[0] = *reinterpret_cast<const uint4*>(p);
  u[1] = *reinterpret_cast<const uint4*>(p + 8);
  return f;
}
// Same A fragment but gathered from an f32 row with on-the-fly bf16 convert.
__device__ __forceinline__ v16bf load_frag_a_f32(const float* p) {
  v16bf f;
#pragma unroll
  for (int i = 0; i < 8; ++i) f[i] = (bf16_t)p[i];
#pragma unroll
  for (int i = 0; i < 8; ++i) f[8 + i] = (bf16_t)p[8 + i];
  return f;
}
// B (32x16 bf16, col-per-lane): lane r holds column n0+r; lanes 0-15 K=k0..k0+15,
// lanes 16-31 K=k0+16..k0+31 (32 contiguous bytes per lane).
__device__ __forceinline__ v16bf load_frag_b(const bf16_t* p) {
  v16bf f;
  uint4* u = reinterpret_cast<uint4*>(&f);
  u[0] = *reinterpret_cast<const uint4*>(p);
  u[1] = *reinterpret_cast<const uint4*>(p + 8);
  return f;
}

__device__ __forceinline__ v8f wmma_bf16(v16bf a, v16bf b, v8f c) {
  return __builtin_amdgcn_wmma_f32_16x16x32_bf16(false, a, false, b, (short)0, c, false, false);
}

// ---------------- generic C[M,N] = A[M,K] * W[N,K]^T + bias ----------------
__global__ __launch_bounds__(256)
void gemm_bias_bf16(const bf16_t* __restrict__ A, int lda,
                    const bf16_t* __restrict__ W, int ldw,
                    const float* __restrict__ bias,
                    float* __restrict__ C, int ldc,
                    int M, int N, int K) {
  const int lane  = threadIdx.x & 31;
  const int wave  = blockIdx.x * (blockDim.x >> 5) + (threadIdx.x >> 5);
  const int ntile = N >> 6;
  const int m0    = (wave / ntile) * 16;
  const int n0    = (wave % ntile) * 64;
  if (m0 >= M) return;
  const int half = lane >> 4;
  const int r    = lane & 15;

  v8f c0 = {}, c1 = {}, c2 = {}, c3 = {};
  const bf16_t* arow = A + (size_t)(m0 + r) * lda + half * 16;
  const bf16_t* w0   = W + (size_t)(n0 +  0 + r) * ldw + half * 16;
  const bf16_t* w1   = W + (size_t)(n0 + 16 + r) * ldw + half * 16;
  const bf16_t* w2   = W + (size_t)(n0 + 32 + r) * ldw + half * 16;
  const bf16_t* w3   = W + (size_t)(n0 + 48 + r) * ldw + half * 16;

  for (int k0 = 0; k0 < K; k0 += 32) {
    v16bf a  = load_frag_a(arow + k0);
    c0 = wmma_bf16(a, load_frag_b(w0 + k0), c0);
    c1 = wmma_bf16(a, load_frag_b(w1 + k0), c1);
    c2 = wmma_bf16(a, load_frag_b(w2 + k0), c2);
    c3 = wmma_bf16(a, load_frag_b(w3 + k0), c3);
  }

  const int rbase = m0 + half * 8;
  v8f acc[4] = {c0, c1, c2, c3};
#pragma unroll
  for (int t = 0; t < 4; ++t) {
    const int gcol = n0 + t * 16 + r;
    const float bv = bias ? bias[gcol] : 0.0f;
#pragma unroll
    for (int j = 0; j < 8; ++j)
      C[(size_t)(rbase + j) * (size_t)ldc + gcol] = acc[t][j] + bv;
  }
}

// ---------------- fused GRU step ----------------
// One wave owns a 16x16 tile of the new hidden state: m-tile (rows/batch) x
// j-tile (hidden cols). It computes all gate projections for that tile with
// WMMA and applies the GRU nonlinearity directly on the accumulators.
// DEC=true : x = emb[tok] gathered in the A-fragment load; gi = x@Wih^T+bih.
// DEC=false: gi comes precomputed (gi_pre, bias included); only gh is computed.
// hb double-buffered (hb_in read as GEMM input across waves, hb_out written).
template <bool DEC>
__global__ __launch_bounds__(256)
void gru_step_k(const bf16_t* __restrict__ hb_in,
                float* __restrict__ hf,
                bf16_t* __restrict__ hb_out,
                const bf16_t* __restrict__ Wih,
                const bf16_t* __restrict__ Whh,
                const float* __restrict__ bih,
                const float* __restrict__ bhh,
                const float* __restrict__ emb,
                const int* __restrict__ tok,
                const float* __restrict__ gi_pre) {
  const int lane = threadIdx.x & 31;
  const int wave = blockIdx.x * (blockDim.x >> 5) + (threadIdx.x >> 5);  // 0..127
  const int mt   = wave >> 5;          // 4 m-tiles
  const int jt   = wave & 31;          // 32 j-tiles (Hn/16)
  const int m0   = mt * 16;
  const int j0   = jt * 16;
  const int half = lane >> 4;
  const int r    = lane & 15;

  const bf16_t* ah_base = hb_in + (size_t)(m0 + r) * Hn + half * 16;
  const float*  ax_base = nullptr;
  if (DEC) {
    const int tk = tok[m0 + r];
    ax_base = emb + (size_t)tk * Hn + half * 16;
  }
  // Whh/Wih rows: [0,H)=r-gate, [H,2H)=z-gate, [2H,3H)=n-gate
  const bf16_t* whr = Whh + (size_t)(0 * Hn + j0 + r) * Hn + half * 16;
  const bf16_t* whz = Whh + (size_t)(1 * Hn + j0 + r) * Hn + half * 16;
  const bf16_t* whn = Whh + (size_t)(2 * Hn + j0 + r) * Hn + half * 16;
  const bf16_t* wir = DEC ? Wih + (size_t)(0 * Hn + j0 + r) * Hn + half * 16 : nullptr;
  const bf16_t* wiz = DEC ? Wih + (size_t)(1 * Hn + j0 + r) * Hn + half * 16 : nullptr;
  const bf16_t* win = DEC ? Wih + (size_t)(2 * Hn + j0 + r) * Hn + half * 16 : nullptr;

  v8f ahr = {}, ahz = {}, ahn = {};
  v8f air = {}, aiz = {}, ain = {};

  for (int k0 = 0; k0 < Hn; k0 += 32) {
    v16bf ah = load_frag_a(ah_base + k0);
    ahr = wmma_bf16(ah, load_frag_b(whr + k0), ahr);
    ahz = wmma_bf16(ah, load_frag_b(whz + k0), ahz);
    ahn = wmma_bf16(ah, load_frag_b(whn + k0), ahn);
    if (DEC) {
      v16bf ax = load_frag_a_f32(ax_base + k0);
      air = wmma_bf16(ax, load_frag_b(wir + k0), air);
      aiz = wmma_bf16(ax, load_frag_b(wiz + k0), aiz);
      ain = wmma_bf16(ax, load_frag_b(win + k0), ain);
    }
  }

  // Epilogue: all six accumulators share (row = m0+half*8+j, col = j0+r).
  const int col   = j0 + r;
  const int rbase = m0 + half * 8;
  const float bhr = bhh[col], bhz = bhh[Hn + col], bhn = bhh[2 * Hn + col];
  float bir = 0.0f, biz = 0.0f, bin = 0.0f;
  if (DEC) { bir = bih[col]; biz = bih[Hn + col]; bin = bih[2 * Hn + col]; }

#pragma unroll
  for (int j = 0; j < 8; ++j) {
    const int row = rbase + j;
    float ir, iz, in_;
    if (DEC) {
      ir = air[j] + bir; iz = aiz[j] + biz; in_ = ain[j] + bin;
    } else {
      const float* g = gi_pre + (size_t)row * H3n;
      ir = g[col]; iz = g[Hn + col]; in_ = g[2 * Hn + col];   // bih already in
    }
    const float hr = ahr[j] + bhr, hz = ahz[j] + bhz, hn = ahn[j] + bhn;
    const float rg = 1.0f / (1.0f + __expf(-(ir + hr)));
    const float zg = 1.0f / (1.0f + __expf(-(iz + hz)));
    const float ng = tanhf(in_ + rg * hn);
    const size_t idx = (size_t)row * Hn + col;
    const float hnew = (1.0f - zg) * ng + zg * hf[idx];
    hf[idx]     = hnew;
    hb_out[idx] = (bf16_t)hnew;
  }
}

// ---------------- elementwise / helper kernels ----------------
__global__ void f32_to_bf16_k(const float* __restrict__ in, bf16_t* __restrict__ out, int n) {
  int i = blockIdx.x * blockDim.x + threadIdx.x;
  if (i < n) out[i] = (bf16_t)in[i];
}

__global__ void init_state_k(float* __restrict__ h, bf16_t* __restrict__ hb, int* __restrict__ tok) {
  int i = blockIdx.x * blockDim.x + threadIdx.x;
  if (i < Bn * Hn) { h[i] = 0.0f; hb[i] = (bf16_t)0.0f; }
  if (i < Bn) tok[i] = 0;   // SOS_IDX
}

// x_all[(t*B + b)*H + j] = emb[src[b*S + t]*H + j]
__global__ void gather_src_all_k(const int* __restrict__ src, const float* __restrict__ emb,
                                 bf16_t* __restrict__ xall) {
  int i = blockIdx.x * blockDim.x + threadIdx.x;       // Sn*Bn*Hn
  int j = i % Hn;
  int rb = i / Hn;
  int t = rb / Bn, b = rb % Bn;
  xall[i] = (bf16_t)emb[(size_t)src[b * Sn + t] * Hn + j];
}

// first-max argmax per row (matches jnp.argmax tie-break)
__global__ void argmax_rows_k(const float* __restrict__ logits, int ldr, int ncol,
                              int* __restrict__ tok) {
  __shared__ float sv[256];
  __shared__ int   si[256];
  const float* row = logits + (size_t)blockIdx.x * ldr;
  float best = -3.4e38f; int bi = 0;
  for (int i = threadIdx.x; i < ncol; i += 256) {
    float v = row[i];
    if (v > best) { best = v; bi = i; }
  }
  sv[threadIdx.x] = best; si[threadIdx.x] = bi;
  __syncthreads();
  for (int s = 128; s > 0; s >>= 1) {
    if (threadIdx.x < s) {
      float ov = sv[threadIdx.x + s]; int oi = si[threadIdx.x + s];
      if (ov > sv[threadIdx.x] || (ov == sv[threadIdx.x] && oi < si[threadIdx.x])) {
        sv[threadIdx.x] = ov; si[threadIdx.x] = oi;
      }
    }
    __syncthreads();
  }
  if (threadIdx.x == 0) tok[blockIdx.x] = si[0];
}

// ---------------- host ----------------
static inline void launch_gemm(const bf16_t* A, int lda, const bf16_t* W, int ldw,
                               const float* bias, float* C, int ldc,
                               int M, int N, int K, hipStream_t s) {
  int waves  = (M / 16) * (N / 64);
  int blocks = waves / 8;
  gemm_bias_bf16<<<blocks, 256, 0, s>>>(A, lda, W, ldw, bias, C, ldc, M, N, K);
}

extern "C" void kernel_launch(void* const* d_in, const int* in_sizes, int n_in,
                              void* d_out, int out_size, void* d_ws, size_t ws_size,
                              hipStream_t stream) {
  const int*   src  = (const int*)d_in[0];
  const float* emb  = (const float*)d_in[1];
  const float* eWih = (const float*)d_in[2];
  const float* eWhh = (const float*)d_in[3];
  const float* ebih = (const float*)d_in[4];
  const float* ebhh = (const float*)d_in[5];
  const float* dWih = (const float*)d_in[6];
  const float* dWhh = (const float*)d_in[7];
  const float* dbih = (const float*)d_in[8];
  const float* dbhh = (const float*)d_in[9];
  const float* fcW  = (const float*)d_in[10];
  const float* fcb  = (const float*)d_in[11];
  float* out = (float*)d_out;

  char* p = (char*)d_ws;
  auto carve = [&](size_t bytes) -> void* {
    void* ret = (void*)p;
    p += (bytes + 255) & ~(size_t)255;
    return ret;
  };
  bf16_t* fcWb   = (bf16_t*)carve((size_t)Vn * Hn * 2);
  bf16_t* eWihb  = (bf16_t*)carve((size_t)H3n * Hn * 2);
  bf16_t* eWhhb  = (bf16_t*)carve((size_t)H3n * Hn * 2);
  bf16_t* dWihb  = (bf16_t*)carve((size_t)H3n * Hn * 2);
  bf16_t* dWhhb  = (bf16_t*)carve((size_t)H3n * Hn * 2);
  bf16_t* embAll = (bf16_t*)carve((size_t)Sn * Bn * Hn * 2);
  float*  giAll  = (float*)carve((size_t)Sn * Bn * H3n * 4);
  float*  hf     = (float*)carve((size_t)Bn * Hn * 4);
  bf16_t* hbA    = (bf16_t*)carve((size_t)Bn * Hn * 2);
  bf16_t* hbB    = (bf16_t*)carve((size_t)Bn * Hn * 2);
  int*    tok    = (int*)carve((size_t)Bn * 4);

  // once-per-launch weight conversion (deterministic)
  f32_to_bf16_k<<<(Vn * Hn + 255) / 256, 256, 0, stream>>>(fcW, fcWb, Vn * Hn);
  f32_to_bf16_k<<<(H3n * Hn + 255) / 256, 256, 0, stream>>>(eWih, eWihb, H3n * Hn);
  f32_to_bf16_k<<<(H3n * Hn + 255) / 256, 256, 0, stream>>>(eWhh, eWhhb, H3n * Hn);
  f32_to_bf16_k<<<(H3n * Hn + 255) / 256, 256, 0, stream>>>(dWih, dWihb, H3n * Hn);
  f32_to_bf16_k<<<(H3n * Hn + 255) / 256, 256, 0, stream>>>(dWhh, dWhhb, H3n * Hn);
  init_state_k<<<(Bn * Hn + 255) / 256, 256, 0, stream>>>(hf, hbA, tok);

  // encoder: batch the input-side matmul across all 64 steps (bih folded in)
  gather_src_all_k<<<(Sn * Bn * Hn) / 256, 256, 0, stream>>>(src, emb, embAll);
  launch_gemm(embAll, Hn, eWihb, Hn, ebih, giAll, H3n, Sn * Bn, H3n, Hn, stream);

  bf16_t* hcur = hbA;
  bf16_t* hnxt = hbB;
  const int gruBlocks = (4 * (Hn / 16)) / 8;   // 128 waves / 8 per block = 16

  // encoder recurrence: one fused node per step
  for (int t = 0; t < Sn; ++t) {
    gru_step_k<false><<<gruBlocks, 256, 0, stream>>>(
        hcur, hf, hnxt, nullptr, eWhhb, nullptr, ebhh, nullptr, nullptr,
        giAll + (size_t)t * Bn * H3n);
    bf16_t* tmp = hcur; hcur = hnxt; hnxt = tmp;
  }

  // decoder: fused GRU (incl. embedding gather) -> fc -> argmax per step
  for (int t = 0; t < Tn; ++t) {
    gru_step_k<true><<<gruBlocks, 256, 0, stream>>>(
        hcur, hf, hnxt, dWihb, dWhhb, dbih, dbhh, emb, tok, nullptr);
    bf16_t* tmp = hcur; hcur = hnxt; hnxt = tmp;
    launch_gemm(hcur, Hn, fcWb, Hn, fcb, out + (size_t)t * Vn, Tn * Vn, Bn, Vn, Hn, stream);
    argmax_rows_k<<<Bn, 256, 0, stream>>>(out + (size_t)t * Vn, Tn * Vn, Vn, tok);
  }
}
